// PPAModel_27934467293572
// MI455X (gfx1250) — matrix operand order repaired and checked by reading, not compiled, not run on hardware
//
#include <hip/hip_runtime.h>
#include <hip/hip_bf16.h>

typedef __attribute__((ext_vector_type(16))) _Float16 v16h;
typedef __attribute__((ext_vector_type(8)))  float    v8f;
typedef __attribute__((ext_vector_type(4)))  unsigned int v4u;

union Frag16 { v16h h; v4u q[2]; };

#define N_NODES 100000
#define N_EDGES 1000000
#define EDGE_BLOCKS 1024

// ---------------------------------------------------------------------------
// Repack a f32 row-major weight [K][N] into f16 WMMA B-fragment layout.
// Fragment (kt,nt) covers K=kt*32..+31, N=nt*16..+15.
// B layout: lane holds column N = lane%16; half h holds
// K = kt*32 + (h&7) + (h>=8)*16 + 8*(lane/16).
// Stored so each lane's 16 halfs (32B) are contiguous.
// ---------------------------------------------------------------------------
__global__ void repack_w(const float* __restrict__ W, unsigned short* __restrict__ out,
                         int K, int N) {
  int t = blockIdx.x * blockDim.x + threadIdx.x;
  int NT = N >> 4, KT = K >> 5;
  int f = t >> 5, lane = t & 31;
  if (f >= KT * NT) return;
  int kt = f / NT, nt = f - kt * NT;
  int n = nt * 16 + (lane & 15);
  int kbase = kt * 32 + (lane >> 4) * 8;
  unsigned short* dst = out + ((size_t)f * 32 + lane) * 16;
  #pragma unroll
  for (int h = 0; h < 16; ++h) {
    _Float16 v = (_Float16)W[(size_t)(kbase + (h & 7) + (h >> 3) * 16) * N + n];
    dst[h] = __builtin_bit_cast(unsigned short, v);
  }
}

// ---------------------------------------------------------------------------
// Node encoder: h = relu(x@W1+b1)@W2+b2, stored as f16 row-major [N_NODES][128].
// One wave per 16-node tile. A-fragment layout (16-bit A 16x32):
// lane holds row M=lane%16; halfs h: K = (h&7) + (h>=8)*16 + 8*(lane/16).
// ---------------------------------------------------------------------------
__global__ __launch_bounds__(128) void encoder_kernel(
    const float* __restrict__ x, const float* __restrict__ b1,
    const float* __restrict__ b2,
    const unsigned short* __restrict__ w1p, const unsigned short* __restrict__ w2p,
    unsigned short* __restrict__ hout) {
  __shared__ _Float16 lds[4][16 * 256];   // 8KB per wave, wave-private
  int warp = threadIdx.x >> 5;
  int lane = threadIdx.x & 31;
  int tile = blockIdx.x * 4 + warp;
  if (tile * 16 >= N_NODES) return;
  int lx = lane & 15, hi = lane >> 4, abase = hi * 8;
  int node0 = tile * 16;
  _Float16* L = &lds[warp][0];

  // A fragments of x (f32 -> f16), K=128 -> 4 k-tiles
  Frag16 a[4];
  #pragma unroll
  for (int kt = 0; kt < 4; ++kt) {
    const float* p = x + (size_t)(node0 + lx) * 128 + kt * 32 + abase;
    #pragma unroll
    for (int j = 0; j < 8; ++j) a[kt].h[j]     = (_Float16)p[j];
    #pragma unroll
    for (int j = 0; j < 8; ++j) a[kt].h[8 + j] = (_Float16)p[16 + j];
  }

  // stage 1: hidden = relu(x@W1 + b1), N=256 -> 16 n-tiles, stage f16 in LDS
  #pragma unroll 1
  for (int nt = 0; nt < 16; ++nt) {
    v8f acc = {};
    #pragma unroll
    for (int kt = 0; kt < 4; ++kt) {
      Frag16 b;
      const v4u* bp = (const v4u*)(w1p + ((size_t)(kt * 16 + nt) * 32 + lane) * 16);
      b.q[0] = bp[0]; b.q[1] = bp[1];
      acc = __builtin_amdgcn_wmma_f32_16x16x32_f16(false, a[kt].h, false, b.h,
                                                   (short)0, acc, false, false);
    }
    int col = nt * 16 + lx;
    float bias = b1[col];
    #pragma unroll
    for (int v = 0; v < 8; ++v) {
      int row = v + hi * 8;
      L[row * 256 + col] = (_Float16)fmaxf(acc[v] + bias, 0.0f);
    }
  }
  asm volatile("s_wait_dscnt 0" ::: "memory");

  // re-stripe hidden into A fragments, K=256 -> 8 k-tiles
  Frag16 ah[8];
  #pragma unroll
  for (int kt = 0; kt < 8; ++kt) {
    const _Float16* p = L + lx * 256 + kt * 32 + abase;
    ah[kt].q[0] = *(const v4u*)(p);
    ah[kt].q[1] = *(const v4u*)(p + 16);
  }
  asm volatile("s_wait_dscnt 0" ::: "memory");

  // stage 2: h = hidden@W2 + b2, N=128 -> 8 n-tiles; stage f16 tile in LDS
  #pragma unroll 1
  for (int nt = 0; nt < 8; ++nt) {
    v8f acc = {};
    #pragma unroll
    for (int kt = 0; kt < 8; ++kt) {
      Frag16 b;
      const v4u* bp = (const v4u*)(w2p + ((size_t)(kt * 8 + nt) * 32 + lane) * 16);
      b.q[0] = bp[0]; b.q[1] = bp[1];
      acc = __builtin_amdgcn_wmma_f32_16x16x32_f16(false, ah[kt].h, false, b.h,
                                                   (short)0, acc, false, false);
    }
    int col = nt * 16 + lx;
    float bias = b2[col];
    #pragma unroll
    for (int v = 0; v < 8; ++v) {
      int row = v + hi * 8;
      L[row * 128 + col] = (_Float16)(acc[v] + bias);
    }
  }
  asm volatile("s_wait_dscnt 0" ::: "memory");

  // coalesced copy of 16x128 f16 tile to global (16B per lane per iter)
  unsigned short* g = hout + (size_t)node0 * 128;
  #pragma unroll
  for (int it = 0; it < 8; ++it) {
    int idx = it * 32 + lane;
    *(v4u*)(g + idx * 8) = *(const v4u*)(L + idx * 8);
  }
}

// ---------------------------------------------------------------------------
// Edge predictor: sigmoid(relu([h_src|h_dst]@W3+b3)@W4 + b4).
// Persistent blocks: each block stages the full packed W3 (128 KB) into LDS
// once, then its 8 waves loop over 32-edge tile-pairs with grid stride.
// B fragments come from LDS (ds_load_b128, overlaps the global A-gather);
// each B fragment feeds two WMMAs (M=32 per wave).
// Layer 2 (256->1) is a VALU FMA on the C layout + shfl_xor reduction.
// ---------------------------------------------------------------------------
__global__ __launch_bounds__(256) void edge_kernel(
    const int* __restrict__ ei, const unsigned short* __restrict__ hf,
    const float* __restrict__ b3, const float* __restrict__ W4,
    const float* __restrict__ b4, const unsigned short* __restrict__ w3p,
    float* __restrict__ out) {
  __shared__ unsigned short w3s[256 * 256];   // 128 KB packed W3 (WGP has 320 KB)
  int tid = threadIdx.x;

  // stage packed W3 into LDS: 32 iters x 256 threads x 16B
  #pragma unroll 1
  for (int it = 0; it < 32; ++it) {
    int idx = (it * 256 + tid) * 8;
    *(v4u*)(w3s + idx) = *(const v4u*)(w3p + idx);
  }
  __syncthreads();

  int warp = tid >> 5, lane = tid & 31;
  int lx = lane & 15, hi = lane >> 4, abase = hi * 8;
  int gw = blockIdx.x * 8 + warp;
  const int nwaves = EDGE_BLOCKS * 8;

  #pragma unroll 1
  for (int w = gw; w < N_EDGES / 32; w += nwaves) {
    int t0 = w * 2;

    // Gather A fragments: e = [h_src(K 0..127) | h_dst(K 128..255)], f16
    Frag16 a[2][8];
    #pragma unroll
    for (int m = 0; m < 2; ++m) {
      int e = (t0 + m) * 16 + lx;
      int s = ei[e];
      int d = ei[N_EDGES + e];
      const unsigned short* ps = hf + (size_t)s * 128;
      const unsigned short* pd = hf + (size_t)d * 128;
      #pragma unroll
      for (int kt = 0; kt < 8; ++kt) {
        const unsigned short* p = (kt < 4) ? (ps + kt * 32) : (pd + (kt - 4) * 32);
        a[m][kt].q[0] = *(const v4u*)(p + abase);
        a[m][kt].q[1] = *(const v4u*)(p + abase + 16);
      }
    }

    v8f acc2a = {}, acc2b = {};
    #pragma unroll 1
    for (int nt = 0; nt < 16; ++nt) {
      v8f c0 = {}, c1 = {};
      #pragma unroll
      for (int kt = 0; kt < 8; ++kt) {
        Frag16 b;
        const v4u* bp = (const v4u*)(w3s + ((kt * 16 + nt) * 32 + lane) * 16);
        b.q[0] = bp[0]; b.q[1] = bp[1];
        c0 = __builtin_amdgcn_wmma_f32_16x16x32_f16(false, a[0][kt].h, false, b.h,
                                                    (short)0, c0, false, false);
        c1 = __builtin_amdgcn_wmma_f32_16x16x32_f16(false, a[1][kt].h, false, b.h,
                                                    (short)0, c1, false, false);
      }
      int col = nt * 16 + lx;
      float bias = b3[col];
      float w4c  = W4[col];
      #pragma unroll
      for (int v = 0; v < 8; ++v) {
        acc2a[v] += fmaxf(c0[v] + bias, 0.0f) * w4c;
        acc2b[v] += fmaxf(c1[v] + bias, 0.0f) * w4c;
      }
    }

    float bb = b4[0];
    #pragma unroll
    for (int v = 0; v < 8; ++v) {
      float s0 = acc2a[v], s1 = acc2b[v];
      s0 += __shfl_xor(s0, 1); s0 += __shfl_xor(s0, 2);
      s0 += __shfl_xor(s0, 4); s0 += __shfl_xor(s0, 8);
      s1 += __shfl_xor(s1, 1); s1 += __shfl_xor(s1, 2);
      s1 += __shfl_xor(s1, 4); s1 += __shfl_xor(s1, 8);
      if (lx == 0) {
        int row = v + hi * 8;
        out[(size_t)(t0 + 0) * 16 + row] = 1.0f / (1.0f + __expf(-(s0 + bb)));
        out[(size_t)(t0 + 1) * 16 + row] = 1.0f / (1.0f + __expf(-(s1 + bb)));
      }
    }
  }
}

// ---------------------------------------------------------------------------
extern "C" void kernel_launch(void* const* d_in, const int* in_sizes, int n_in,
                              void* d_out, int out_size, void* d_ws, size_t ws_size,
                              hipStream_t stream) {
  const float* x  = (const float*)d_in[0];
  const int*   ei = (const int*)  d_in[1];
  const float* W1 = (const float*)d_in[2];
  const float* b1 = (const float*)d_in[3];
  const float* W2 = (const float*)d_in[4];
  const float* b2 = (const float*)d_in[5];
  const float* W3 = (const float*)d_in[6];
  const float* b3 = (const float*)d_in[7];
  const float* W4 = (const float*)d_in[8];
  const float* b4 = (const float*)d_in[9];
  float* out = (float*)d_out;

  // workspace layout (f16 halves as ushort)
  unsigned short* hf  = (unsigned short*)d_ws;          // 100000*128 = 25.6 MB
  unsigned short* w1p = hf  + (size_t)N_NODES * 128;    // 128x256 -> 64 KB
  unsigned short* w2p = w1p + 128 * 256;                // 256x128 -> 64 KB
  unsigned short* w3p = w2p + 256 * 128;                // 256x256 -> 128 KB

  // repack weights into WMMA B-fragment layout (f32 -> f16)
  repack_w<<<8,  256, 0, stream>>>(W1, w1p, 128, 256);  // 64 frags * 32 lanes
  repack_w<<<8,  256, 0, stream>>>(W2, w2p, 256, 128);  // 64 frags
  repack_w<<<16, 256, 0, stream>>>(W3, w3p, 256, 256);  // 128 frags

  // node encoder: 6250 tiles of 16 nodes, 4 waves/block
  encoder_kernel<<<(N_NODES / 16 + 3) / 4, 128, 0, stream>>>(x, b1, b2, w1p, w2p, hf);

  // edge predictor: persistent blocks, W3 cached in LDS
  edge_kernel<<<EDGE_BLOCKS, 256, 0, stream>>>(ei, hf, b3, W4, b4, w3p, out);
}
